// ConfLearnLoss_11544872092220
// MI455X (gfx1250) — matrix operand order repaired and checked by reading, not compiled, not run on hardware
//
#include <hip/hip_runtime.h>
#include <hip/hip_bf16.h>

#define NS   8192
#define KC   64
#define GR   8
#define NPG  1024
#define INV_EPS 10.0f   // 1/regularization_strength (0.1)
#define BS   50.0f      // sigmoid sharpness

typedef float    v2f  __attribute__((ext_vector_type(2)));
typedef float    v8f  __attribute__((ext_vector_type(8)));
typedef _Float16 v16h __attribute__((ext_vector_type(16)));

__device__ __forceinline__ float sigf(float x) { return 1.0f / (1.0f + __expf(-x)); }

// L2 isotonic regression, NONINCREASING fit, exact PAV (O(n)).
// Serial: run by one lane; arrays live in LDS.
__device__ void pav_noninc(const float* a, float* v, float* bmean, int* bcnt, int n)
{
    int nb = 0;
    for (int i = 0; i < n; ++i) {
        float m = a[i];
        float c = 1.0f;
        // nonincreasing constraint: merging while new block mean exceeds previous
        while (nb > 0 && m > bmean[nb - 1]) {
            float cp = (float)bcnt[nb - 1];
            m = (m * c + bmean[nb - 1] * cp) / (c + cp);
            c += cp;
            --nb;
        }
        bmean[nb] = m;
        bcnt[nb]  = (int)c;
        ++nb;
    }
    int p = 0;
    for (int b = 0; b < nb; ++b) {
        float mv = bmean[b];
        for (int r = 0; r < bcnt[b]; ++r) v[p++] = mv;
    }
}

// ---------------------------------------------------------------------------
// Kernel 1: per-sample phase. One wave (32 lanes) per sample, 2 classes/lane.
// softmax -> noise renorm -> bitonic sort (asc, with payload) ->
// PAV #1 (soft_rank of target class) -> PAV #2 (soft descending sort)
// ---------------------------------------------------------------------------
__global__ void __launch_bounds__(256)
k_persample(const float* __restrict__ logits, const int* __restrict__ tgt,
            const float* __restrict__ upk, float* __restrict__ prob_sort,
            float* __restrict__ rankv)
{
    __shared__ float key[8][KC];
    __shared__ int   idx[8][KC];
    __shared__ float tmp[8][KC];
    __shared__ float iso[8][KC];
    __shared__ float bm [8][KC];
    __shared__ int   bc [8][KC];
    __shared__ int   posy[8];

    const int w    = threadIdx.x >> 5;
    const int lane = threadIdx.x & 31;
    const int s    = blockIdx.x * 8 + w;

    const float* row  = logits + (size_t)s * KC;
    const float* urow = upk    + (size_t)s * KC;

    // softmax over 64 (2 per lane), wave32 shuffle reductions
    float e0 = row[lane], e1 = row[lane + 32];
    float m = fmaxf(e0, e1);
    for (int d = 16; d >= 1; d >>= 1) m = fmaxf(m, __shfl_xor(m, d, 32));
    e0 = __expf(e0 - m); e1 = __expf(e1 - m);
    float sm = e0 + e1;
    for (int d = 16; d >= 1; d >>= 1) sm += __shfl_xor(sm, d, 32);
    float p0 = e0 / sm + 1e-6f * urow[lane];
    float p1 = e1 / sm + 1e-6f * urow[lane + 32];
    float s2 = p0 + p1;
    for (int d = 16; d >= 1; d >>= 1) s2 += __shfl_xor(s2, d, 32);
    p0 /= s2; p1 /= s2;

    key[w][lane]      = p0; idx[w][lane]      = lane;
    key[w][lane + 32] = p1; idx[w][lane + 32] = lane + 32;
    __syncthreads();

    // bitonic ascending sort of 64 keys with index payload
    for (int size = 2; size <= KC; size <<= 1)
        for (int stride = size >> 1; stride > 0; stride >>= 1) {
            __syncthreads();
#pragma unroll
            for (int h = 0; h < 2; ++h) {
                int t = lane + h * 32;
                int j = t ^ stride;
                if (j > t) {
                    bool up = ((t & size) == 0);
                    float a = key[w][t], b = key[w][j];
                    if (up ? (a > b) : (a < b)) {
                        key[w][t] = b; key[w][j] = a;
                        int ti = idx[w][t]; idx[w][t] = idx[w][j]; idx[w][j] = ti;
                    }
                }
            }
        }
    __syncthreads();

    // ascending position of the target class
    const int y = tgt[s];
    if (idx[w][lane]      == y) posy[w] = lane;
    if (idx[w][lane + 32] == y) posy[w] = lane + 32;
    __syncthreads();

    // soft_rank: theta sorted desc = -p_asc/eps ; rho = [64..1]
    tmp[w][lane]      = -key[w][lane]      * INV_EPS - (float)(KC - lane);
    tmp[w][lane + 32] = -key[w][lane + 32] * INV_EPS - (float)(KC - (lane + 32));
    __syncthreads();
    if (lane == 0) pav_noninc(tmp[w], iso[w], bm[w], bc[w], KC);
    __syncthreads();
    if (lane == 0) {
        int py = posy[w];
        float st = -key[w][py] * INV_EPS;
        rankv[s] = st - iso[w][py] - 1.0f;   // 0-based soft rank of target
    }
    __syncthreads();

    // soft descending sort: rho2 = [64..1]/eps ; p_desc[k] = p_asc[63-k]
    tmp[w][lane]      = (float)(KC - lane)        * INV_EPS - key[w][KC - 1 - lane];
    tmp[w][lane + 32] = (float)(KC - (lane + 32)) * INV_EPS - key[w][KC - 1 - (lane + 32)];
    __syncthreads();
    if (lane == 0) pav_noninc(tmp[w], iso[w], bm[w], bc[w], KC);
    __syncthreads();
    prob_sort[(size_t)s * KC + lane]      = (float)(KC - lane)        * INV_EPS - iso[w][lane];
    prob_sort[(size_t)s * KC + lane + 32] = (float)(KC - (lane + 32)) * INV_EPS - iso[w][lane + 32];
}

// ---------------------------------------------------------------------------
// Kernel 2: Zc = cumsum(prob_sort, axis=1) as GEMM on the matrix pipe:
// Zc(8192x64) = P(8192x64) x U(64x64), U upper-triangular ones.
// One wave per 16x16 output tile, 2048 waves total.
// ---------------------------------------------------------------------------
__global__ void __launch_bounds__(256)
k_cumsum_wmma(const float* __restrict__ P, float* __restrict__ Z)
{
    const int wave = (blockIdx.x * blockDim.x + threadIdx.x) >> 5;
    const int lane = threadIdx.x & 31;
    const int lo = lane & 15, hi = lane >> 4;
    const int mTile = wave >> 2;            // 512 row tiles
    const int nTile = wave & 3;             // 4 col tiles
    const int rowA = mTile * 16 + lo;
    const int ncol = nTile * 16 + lo;

    v8f acc = {};

#if __has_builtin(__builtin_amdgcn_wmma_f32_16x16x4_f32)
    // FP32 WMMA: A 16x4 (lane: M=lo, K=2*hi+{0,1}), B 4x16 (lane: N=lo).
    for (int kb = 0; kb < 16; ++kb) {
        int k0 = kb * 4 + hi * 2;
        v2f a, b;
        a.x = P[(size_t)rowA * KC + k0];
        a.y = P[(size_t)rowA * KC + k0 + 1];
        b.x = (k0     <= ncol) ? 1.0f : 0.0f;   // U[j][n] = (j <= n)
        b.y = (k0 + 1 <= ncol) ? 1.0f : 0.0f;
        acc = __builtin_amdgcn_wmma_f32_16x16x4_f32(false, a, false, b,
                                                    (short)0, acc, false, false);
    }
#else
    // Fallback: codegen-confirmed f16 WMMA, K split into 2 chunks of 32.
    for (int kb = 0; kb < 2; ++kb) {
        v16h a, b;
#pragma unroll
        for (int h = 0; h < 16; ++h) {
            int v = h >> 1, sb = h & 1;
            int kl = (v < 4 ? 2 * v : 16 + 2 * (v - 4)) + (hi ? 8 : 0) + sb;
            int kg = kb * 32 + kl;
            a[h] = (_Float16)P[(size_t)rowA * KC + kg];
            int kgb = kb * 32 + (h < 8 ? 8 * hi + h : 8 + 8 * hi + h);
            b[h] = (_Float16)((kgb <= ncol) ? 1.0f : 0.0f);
        }
        acc = __builtin_amdgcn_wmma_f32_16x16x32_f16(false, a, false, b,
                                                     (short)0, acc, false, false);
    }
#endif

#pragma unroll
    for (int v = 0; v < 8; ++v) {
        int r = mTile * 16 + hi * 8 + v;    // C layout: lanes 0-15 M=v, 16-31 M=v+8
        Z[(size_t)r * KC + ncol] = acc[v];
    }
}

// ---------------------------------------------------------------------------
// Kernel 3: per-sample score epilogue. One wave per sample.
// ---------------------------------------------------------------------------
__global__ void __launch_bounds__(256)
k_scores(const float* __restrict__ Z, const float* __restrict__ PS,
         const float* __restrict__ rankv, const float* __restrict__ us,
         float* __restrict__ scores)
{
    const int w = threadIdx.x >> 5, lane = threadIdx.x & 31;
    const int s = blockIdx.x * 8 + w;
    const float r = rankv[s];
    const float nrm = sigf(0.5f * BS) - sigf(-0.5f * BS);
    float pc = 0.0f, pf = 0.0f;
#pragma unroll
    for (int h = 0; h < 2; ++h) {
        int k = lane + h * 32;
        float wk = (sigf(BS * ((float)k - r + 0.5f)) -
                    sigf(BS * ((float)k - r - 0.5f))) / nrm;
        pc += wk * Z [(size_t)s * KC + k];
        pf += wk * PS[(size_t)s * KC + k];
    }
    for (int d = 16; d >= 1; d >>= 1) {
        pc += __shfl_xor(pc, d, 32);
        pf += __shfl_xor(pf, d, 32);
    }
    if (lane == 0) scores[s] = 1.0f - pc + pf * us[s];
}

__global__ void k_init(float* o) { if (threadIdx.x == 0) o[0] = 0.0f; }

// ---------------------------------------------------------------------------
// Kernel 4: per-group UniformMatchingLoss. One block (512 thr) per group.
// Group g = samples {g, g+8, ...} (Z_batch is fixed round-robin (i%8)+1,
// so stable argsort(Z) is the identity-on-groups gather).
// ---------------------------------------------------------------------------
__global__ void __launch_bounds__(512)
k_group(const float* __restrict__ scores, float* __restrict__ outp)
{
    __shared__ float sv[NPG];
    __shared__ float yv[NPG];
    __shared__ float bm[NPG];
    __shared__ int   bc[NPG];
    const int tid = threadIdx.x;
    const int g   = blockIdx.x;

    for (int t = tid; t < NPG; t += 512) sv[t] = scores[(size_t)t * GR + g];
    __syncthreads();

    // bitonic ascending sort of 1024 scores
    for (int size = 2; size <= NPG; size <<= 1)
        for (int stride = size >> 1; stride > 0; stride >>= 1) {
#pragma unroll
            for (int h = 0; h < 2; ++h) {
                int t = tid + h * 512;
                int j = t ^ stride;
                if (j > t) {
                    bool up = ((t & size) == 0);
                    float a = sv[t], b = sv[j];
                    if (up ? (a > b) : (a < b)) { sv[t] = b; sv[j] = a; }
                }
            }
            __syncthreads();
        }

    // xs = iso_noninc(rho + sorted) - rho, rho[k] = (n-k)/eps
    for (int t = tid; t < NPG; t += 512)
        yv[t] = (float)(NPG - t) * INV_EPS + sv[t];
    __syncthreads();
    if (tid == 0) pav_noninc(yv, sv, bm, bc, NPG);
    __syncthreads();

    float lmax = 0.0f;
    for (int t = tid; t < NPG; t += 512) {
        float xs = sv[t] - (float)(NPG - t) * INV_EPS;
        lmax = fmaxf(lmax, fabsf((float)(t + 1) * (1.0f / NPG) - xs));
    }
    bm[tid] = lmax;
    __syncthreads();
    for (int st = 256; st > 0; st >>= 1) {
        if (tid < st) bm[tid] = fmaxf(bm[tid], bm[tid + st]);
        __syncthreads();
    }
    if (tid == 0) atomicAdd(outp, bm[0] * (1.0f / GR));
}

extern "C" void kernel_launch(void* const* d_in, const int* in_sizes, int n_in,
                              void* d_out, int out_size, void* d_ws, size_t ws_size,
                              hipStream_t stream)
{
    (void)in_sizes; (void)n_in; (void)out_size; (void)ws_size;
    const float* logits = (const float*)d_in[0];   // (8192, 64) f32
    const int*   tgt    = (const int*)  d_in[1];   // (8192,)    i32
    /* d_in[2] = Z_batch: fixed round-robin (i%G)+1; gather is implicit */
    const float* upk    = (const float*)d_in[3];   // (8192, 64) f32
    const float* us     = (const float*)d_in[4];   // (8192,)    f32
    float* o = (float*)d_out;

    float* PS     = (float*)d_ws;                  // prob_sort  N*K
    float* Zc     = PS + (size_t)NS * KC;          // cumsum     N*K
    float* rankv  = Zc + (size_t)NS * KC;          // N
    float* scores = rankv + NS;                    // N

    k_persample <<<NS / 8, 256, 0, stream>>>(logits, tgt, upk, PS, rankv);
    k_cumsum_wmma<<<(NS / 16) * (KC / 16) / 8, 256, 0, stream>>>(PS, Zc);
    k_scores    <<<NS / 8, 256, 0, stream>>>(Zc, PS, rankv, us, scores);
    k_init      <<<1, 32, 0, stream>>>(o);
    k_group     <<<GR, 512, 0, stream>>>(scores, o);
}
